// Model_45423574123024
// MI455X (gfx1250) — matrix operand order repaired and checked by reading, not compiled, not run on hardware
//
#include <hip/hip_runtime.h>
#include <hip/hip_bf16.h>

// ---------------- CDNA5 WMMA types ----------------
typedef __attribute__((ext_vector_type(16))) __bf16 v16bf;
typedef __attribute__((ext_vector_type(8)))  float  v8f;

static __device__ __forceinline__ float sigf(float x) { return 1.0f / (1.0f + __expf(-x)); }

// =====================================================================
// Weight pack: logical B[k][n] (k over [Wih | Whh | pad], n = output col)
// packed into WMMA-B fragment order: block(kt,nt) of 512 bf16, lane-major.
// Lane L: n = nt*16 + (L&15); k = kt*32 + (L>>4)*16 + e, e=0..15 contiguous.
// =====================================================================
__global__ void pack_B_k(const float* __restrict__ Wih, const float* __restrict__ Whh,
                         __bf16* __restrict__ Bp, int in_dim, int hid, int Ktiles) {
  int idx = blockIdx.x * blockDim.x + threadIdx.x;
  int total = Ktiles * 64 * 512;
  if (idx >= total) return;
  int e    = idx & 15;
  int lane = (idx >> 4) & 31;
  int blk  = idx >> 9;
  int nt = blk % 64, kt = blk / 64;
  int n = nt * 16 + (lane & 15);
  int k = kt * 32 + (lane >> 4) * 16 + e;
  float val = 0.f;
  if (k < in_dim)            val = Wih[(size_t)n * in_dim + k];
  else if (k < in_dim + hid) val = Whh[(size_t)n * hid + (k - in_dim)];
  Bp[idx] = (__bf16)val;
}

__global__ void bias_k(const float* __restrict__ bih, const float* __restrict__ bhh,
                       float* __restrict__ b) {
  int i = blockIdx.x * blockDim.x + threadIdx.x;
  if (i < 1024) b[i] = bih[i] + bhh[i];
}

// =====================================================================
// A pack: logical A row m = [X1 | X2 | H | pad], packed into WMMA-A
// fragment order (contiguous k per lane half, same as B pack). Block(mt,kt):
// lane L: m = mt*16 + (L&15), khalf = L>>4;
//   k = khalf*16 + e, e=0..15 contiguous
// =====================================================================
__global__ void pack_A_k(const float* __restrict__ X1, size_t ld1, int d1,
                         const float* __restrict__ X2, size_t ld2, int d2,
                         const float* __restrict__ Hst, int hld,
                         __bf16* __restrict__ A, int Mv, int Mtiles, int hid, int Ktiles) {
  int idx = blockIdx.x * blockDim.x + threadIdx.x;
  int total = Mtiles * Ktiles * 512;
  if (idx >= total) return;
  int e    = idx & 15;
  int lane = (idx >> 4) & 31;
  int blk  = idx >> 9;
  int kt = blk % Ktiles, mt = blk / Ktiles;
  int m = mt * 16 + (lane & 15);
  int khalf = lane >> 4;
  int kin = khalf * 16 + e;
  int k = kt * 32 + kin;
  float val = 0.f;
  if (m < Mv) {
    if (k < d1)                     val = X1[(size_t)m * ld1 + k];
    else if (k < d1 + d2)           val = X2[(size_t)m * ld2 + (k - d1)];
    else if (k < d1 + d2 + hid)     val = Hst[(size_t)m * hld + (k - d1 - d2)];
  }
  A[idx] = (__bf16)val;
}

// =====================================================================
// GEMM: C(M x 1024) = A(M x Kpad) * B(Kpad x 1024) + bias,  bf16 in / f32 acc.
// One wave = 2x4 super-tile (2 M-tiles x 4 N-tiles, 8 accumulators):
// A fragments reused 4x, B fragments 2x -> ~2.7x less L2 traffic than 1x1.
// __launch_bounds__(256,1): lift VGPR budget so the 8 f32 accumulators
// (64 VGPRs) + 6 fragments (48 VGPRs) never spill to scratch.
// Prefetch is unconditional (speculative, drop-on-fault; one-past-end
// addresses stay inside the contiguous workspace) to keep the loop
// straight-line.
// =====================================================================
__global__ void __launch_bounds__(256, 1)
gemm_k(const __bf16* __restrict__ A, const __bf16* __restrict__ Bm,
       const float* __restrict__ bias, float* __restrict__ C,
       int Mtiles, int Ntiles, int Ktiles) {
  int gw   = (blockIdx.x * blockDim.x + threadIdx.x) >> 5;
  int lane = threadIdx.x & 31;
  int Mt2 = (Mtiles + 1) >> 1;
  int Nt4 = Ntiles >> 2;
  if (gw >= Mt2 * Nt4) return;
  int mtb = (gw / Nt4) * 2;
  int ntb = (gw % Nt4) * 4;
  bool two = (mtb + 1) < Mtiles;      // wave-uniform

  v8f acc[8];
#pragma unroll
  for (int i = 0; i < 8; ++i) acc[i] = (v8f){0.f, 0.f, 0.f, 0.f, 0.f, 0.f, 0.f, 0.f};

  const __bf16* Ap0 = A + ((size_t)mtb * Ktiles) * 512 + lane * 16;
  const __bf16* Ap1 = Ap0 + (size_t)Ktiles * 512;     // garbage if !two; never stored
  const __bf16* Bp0 = Bm + (size_t)ntb * 512 + lane * 16;

  for (int kt = 0; kt < Ktiles; ++kt) {
    const __bf16* bp = Bp0 + (size_t)kt * Ntiles * 512;
    // next K-tile is 64KB away in B: pull it toward L2 ahead of the waits
    __builtin_prefetch(Ap0 + (size_t)(kt + 1) * 512, 0, 1);
    __builtin_prefetch(bp + (size_t)Ntiles * 512, 0, 1);
    v16bf a0 = *(const v16bf*)(Ap0 + (size_t)kt * 512);
    v16bf a1 = *(const v16bf*)(Ap1 + (size_t)kt * 512);
    v16bf b0 = *(const v16bf*)(bp);
    v16bf b1 = *(const v16bf*)(bp + 512);
    v16bf b2 = *(const v16bf*)(bp + 1024);
    v16bf b3 = *(const v16bf*)(bp + 1536);
    acc[0] = __builtin_amdgcn_wmma_f32_16x16x32_bf16(false, a0, false, b0, (short)0, acc[0], false, false);
    acc[1] = __builtin_amdgcn_wmma_f32_16x16x32_bf16(false, a0, false, b1, (short)0, acc[1], false, false);
    acc[2] = __builtin_amdgcn_wmma_f32_16x16x32_bf16(false, a0, false, b2, (short)0, acc[2], false, false);
    acc[3] = __builtin_amdgcn_wmma_f32_16x16x32_bf16(false, a0, false, b3, (short)0, acc[3], false, false);
    acc[4] = __builtin_amdgcn_wmma_f32_16x16x32_bf16(false, a1, false, b0, (short)0, acc[4], false, false);
    acc[5] = __builtin_amdgcn_wmma_f32_16x16x32_bf16(false, a1, false, b1, (short)0, acc[5], false, false);
    acc[6] = __builtin_amdgcn_wmma_f32_16x16x32_bf16(false, a1, false, b2, (short)0, acc[6], false, false);
    acc[7] = __builtin_amdgcn_wmma_f32_16x16x32_bf16(false, a1, false, b3, (short)0, acc[7], false, false);
  }

  int nlo = lane & 15;
  int mrow0 = mtb * 16 + (lane >> 4) * 8;
#pragma unroll
  for (int j = 0; j < 4; ++j) {
    int col = (ntb + j) * 16 + nlo;
    float bv = bias ? bias[col] : 0.f;
#pragma unroll
    for (int r = 0; r < 8; ++r)
      C[(size_t)(mrow0 + r) * 1024 + col] = acc[j][r] + bv;
  }
  if (two) {
    int mrow1 = (mtb + 1) * 16 + (lane >> 4) * 8;
#pragma unroll
    for (int j = 0; j < 4; ++j) {
      int col = (ntb + j) * 16 + nlo;
      float bv = bias ? bias[col] : 0.f;
#pragma unroll
      for (int r = 0; r < 8; ++r)
        C[(size_t)(mrow1 + r) * 1024 + col] = acc[4 + j][r] + bv;
    }
  }
}

// =====================================================================
// Fused LSTM cell: gates (M x 1024 as [i|f|g|o]) -> h,c. Optional extra
// output: unconditional (pyramid hidden trace) or conditional on length
// (encoder last-hidden gather).
// =====================================================================
__global__ void lstm_cell_k(const float* __restrict__ gates, float* __restrict__ h,
                            float* __restrict__ c, float* __restrict__ extra,
                            const int* __restrict__ lens, int t, int Mv) {
  int idx = blockIdx.x * blockDim.x + threadIdx.x;
  if (idx >= Mv * 256) return;
  int row = idx >> 8, col = idx & 255;
  const float* g = gates + (size_t)row * 1024;
  float ig = sigf(g[col]);
  float fg = sigf(g[col + 256]);
  float gg = tanhf(g[col + 512]);
  float og = sigf(g[col + 768]);
  float cn = fg * c[idx] + ig * gg;
  float hn = og * tanhf(cn);
  c[idx] = cn;
  h[idx] = hn;
  if (extra) {
    if (lens) { if (lens[row] - 1 == t) extra[idx] = hn; }
    else extra[idx] = hn;
  }
}

__global__ void zero_k(float* __restrict__ p, int n) {
  int i = blockIdx.x * blockDim.x + threadIdx.x;
  if (i < n) p[i] = 0.f;
}

// hlast (2880 x 256) -> seq1 (8 x 180 x 512)
__global__ void reshape_hlast_k(const float* __restrict__ hlast, float* __restrict__ seq) {
  int idx = blockIdx.x * blockDim.x + threadIdx.x;
  if (idx >= 8 * 180 * 512) return;
  int f = idx & 511;
  int rest = idx >> 9;
  int l = rest % 180, b = rest / 180;
  int n = b * 360 + l * 2 + (f >= 256 ? 1 : 0);
  seq[idx] = hlast[(size_t)n * 256 + (f & 255)];
}

// hsf/hsb (L x 8 x 256) -> next seq (8 x L/2 x 512): avg dirs, halve time.
__global__ void pyr_combine_k(const float* __restrict__ hsf, const float* __restrict__ hsb,
                              float* __restrict__ outSeq, int L) {
  int half = L / 2;
  int total = 8 * half * 512;
  int idx = blockIdx.x * blockDim.x + threadIdx.x;
  if (idx >= total) return;
  int f = idx & 511;
  int rest = idx >> 9;
  int j = rest % half, b = rest / half;
  int t = 2 * j + (f >= 256 ? 1 : 0);
  int c = f & 255;
  size_t o = (size_t)(t * 8 + b) * 256 + c;
  outSeq[idx] = 0.5f * (hsf[o] + hsb[o]);
}

// Final stage: avg dirs only -> (8 x L x 256)
__global__ void avg_combine_k(const float* __restrict__ hsf, const float* __restrict__ hsb,
                              float* __restrict__ out, int L) {
  int total = 8 * L * 256;
  int idx = blockIdx.x * blockDim.x + threadIdx.x;
  if (idx >= total) return;
  int c = idx & 255;
  int rest = idx >> 8;
  int t = rest % L, b = rest / L;
  size_t o = (size_t)(t * 8 + b) * 256 + c;
  out[(size_t)(b * L + t) * 256 + c] = 0.5f * (hsf[o] + hsb[o]);
}

// LayerNorm(256) + tanh(W1 x + b1), one block per (b,t) row.
__global__ void ln_mlp_k(const float* __restrict__ X, const float* __restrict__ g,
                         const float* __restrict__ bta, const float* __restrict__ W1,
                         const float* __restrict__ b1, float* __restrict__ Y) {
  __shared__ float sx[256];
  __shared__ float red[256];
  int r = blockIdx.x, tid = threadIdx.x;
  float v = X[(size_t)r * 256 + tid];
  red[tid] = v;
  __syncthreads();
  for (int s = 128; s > 0; s >>= 1) { if (tid < s) red[tid] += red[tid + s]; __syncthreads(); }
  float mu = red[0] * (1.f / 256.f);
  __syncthreads();
  float d = v - mu;
  red[tid] = d * d;
  __syncthreads();
  for (int s = 128; s > 0; s >>= 1) { if (tid < s) red[tid] += red[tid + s]; __syncthreads(); }
  float var = red[0] * (1.f / 256.f);
  float xn = d * rsqrtf(var + 1e-5f) * g[tid] + bta[tid];
  __syncthreads();
  sx[tid] = xn;
  __syncthreads();
  float acc = b1[tid];
  const float* w = W1 + (size_t)tid * 256;
#pragma unroll 4
  for (int k = 0; k < 256; ++k) acc += w[k] * sx[k];
  Y[(size_t)r * 256 + tid] = tanhf(acc);
}

// Decoder attention step: si=tanh(h1 W2^T + b2); softmax(si . Henc); context;
// yi = softmax([si,ci] Wo^T + bo). One block, 256 threads.
__global__ void dec_attn_k(const float* __restrict__ h1, const float* __restrict__ W2,
                           const float* __restrict__ b2, const float* __restrict__ Henc,
                           const float* __restrict__ Wo, const float* __restrict__ bo,
                           float* __restrict__ yi, float* __restrict__ ci,
                           float* __restrict__ ys_step) {
  __shared__ float si[8 * 256];
  __shared__ float sc[8 * 45];
  __shared__ float sci[8 * 256];
  __shared__ float slog[64];
  int tid = threadIdx.x;
  for (int o = tid; o < 2048; o += 256) {
    int b = o >> 8, hh = o & 255;
    float a = b2[hh];
    const float* hb = h1 + b * 256;
    const float* w = W2 + (size_t)hh * 256;
    for (int k = 0; k < 256; ++k) a += hb[k] * w[k];
    si[o] = tanhf(a);
  }
  __syncthreads();
  for (int o = tid; o < 360; o += 256) {
    int b = o / 45, l = o % 45;
    const float* m = Henc + (size_t)(b * 45 + l) * 256;
    const float* s = si + b * 256;
    float a = 0.f;
    for (int k = 0; k < 256; ++k) a += s[k] * m[k];
    sc[o] = a;
  }
  __syncthreads();
  if (tid < 8) {
    float mx = -1e30f;
    for (int l = 0; l < 45; ++l) mx = fmaxf(mx, sc[tid * 45 + l]);
    float sm = 0.f;
    for (int l = 0; l < 45; ++l) { float e = __expf(sc[tid * 45 + l] - mx); sc[tid * 45 + l] = e; sm += e; }
    float inv = 1.f / sm;
    for (int l = 0; l < 45; ++l) sc[tid * 45 + l] *= inv;
  }
  __syncthreads();
  for (int o = tid; o < 2048; o += 256) {
    int b = o >> 8, hh = o & 255;
    float a = 0.f;
    for (int l = 0; l < 45; ++l) a += sc[b * 45 + l] * Henc[(size_t)(b * 45 + l) * 256 + hh];
    sci[o] = a;
    ci[o] = a;
  }
  __syncthreads();
  for (int o = tid; o < 64; o += 256) {
    int b = o >> 3, j = o & 7;
    float a = bo[j];
    const float* w = Wo + (size_t)j * 512;
    for (int k = 0; k < 256; ++k) a += si[b * 256 + k] * w[k];
    for (int k = 0; k < 256; ++k) a += sci[b * 256 + k] * w[256 + k];
    slog[o] = a;
  }
  __syncthreads();
  if (tid < 8) {
    int b = tid;
    float mx = -1e30f;
    for (int j = 0; j < 8; ++j) mx = fmaxf(mx, slog[b * 8 + j]);
    float e[8], sm = 0.f;
    for (int j = 0; j < 8; ++j) { e[j] = __expf(slog[b * 8 + j] - mx); sm += e[j]; }
    float inv = 1.f / sm;
    for (int j = 0; j < 8; ++j) { float y = e[j] * inv; yi[b * 8 + j] = y; ys_step[b * 8 + j] = y; }
  }
}

// ys (30 x 8 x 8) -> out (8*30, 8) with log
__global__ void final_k(const float* __restrict__ ys, float* __restrict__ out) {
  int idx = blockIdx.x * blockDim.x + threadIdx.x;
  if (idx >= 1920) return;
  int j = idx & 7;
  int rest = idx >> 3;
  int s = rest % 30, b = rest / 30;
  out[idx] = __logf(ys[(size_t)(s * 8 + b) * 8 + j]);
}

// =====================================================================
extern "C" void kernel_launch(void* const* d_in, const int* in_sizes, int n_in,
                              void* d_out, int out_size, void* d_ws, size_t ws_size,
                              hipStream_t stream) {
  (void)in_sizes; (void)n_in; (void)out_size; (void)ws_size;
  constexpr int Nn = 2880, Tt = 48, FEAT = 43;

  const float* xs       = (const float*)d_in[0];
  const int*   xlengths = (const int*)d_in[1];
  struct Lw { const float *Wih, *Whh, *bih, *bhh; };
  auto lw = [&](int base) { return Lw{(const float*)d_in[base], (const float*)d_in[base + 1],
                                      (const float*)d_in[base + 2], (const float*)d_in[base + 3]}; };
  Lw enc = lw(2);
  Lw pyr[6]; for (int i = 0; i < 6; ++i) pyr[i] = lw(6 + i * 4);
  const float* gamma = (const float*)d_in[30];
  const float* beta  = (const float*)d_in[31];
  const float* W1 = (const float*)d_in[32]; const float* b1 = (const float*)d_in[33];
  const float* W2 = (const float*)d_in[34]; const float* b2 = (const float*)d_in[35];
  Lw dec0 = lw(36), dec1 = lw(40);
  const float* Wo = (const float*)d_in[44]; const float* bo = (const float*)d_in[45];
  float* out = (float*)d_out;

  // ---- workspace carve ----
  char* ws = (char*)d_ws;
  size_t cur = 0;
  auto carve = [&](size_t bytes) -> char* {
    char* p = ws + cur; cur = (cur + bytes + 255) & ~(size_t)255; return p;
  };
  __bf16* encB = (__bf16*)carve(10 * 64 * 512 * 2);        float* encBias = (float*)carve(1024 * 4);
  __bf16* pyrB[6]; float* pyrBias[6];
  for (int i = 0; i < 6; ++i) { pyrB[i] = (__bf16*)carve(24 * 64 * 512 * 2); pyrBias[i] = (float*)carve(1024 * 4); }
  __bf16* dec0B = (__bf16*)carve(17 * 64 * 512 * 2);       float* dec0Bias = (float*)carve(1024 * 4);
  __bf16* dec1B = (__bf16*)carve(16 * 64 * 512 * 2);       float* dec1Bias = (float*)carve(1024 * 4);
  __bf16* Abf  = (__bf16*)carve((size_t)180 * 10 * 512 * 2);
  float* gates = (float*)carve((size_t)Nn * 1024 * 4);
  float* hEnc  = (float*)carve((size_t)Nn * 256 * 4);
  float* cEnc  = (float*)carve((size_t)Nn * 256 * 4);
  float* hlast = (float*)carve((size_t)Nn * 256 * 4);
  float* seq1  = (float*)carve((size_t)8 * 180 * 512 * 4);
  float* seq2  = (float*)carve((size_t)8 * 90 * 512 * 4);
  float* seq3  = (float*)carve((size_t)8 * 45 * 512 * 4);
  float* hsF   = (float*)carve((size_t)180 * 8 * 256 * 4);
  float* hsB   = (float*)carve((size_t)180 * 8 * 256 * 4);
  float* hSt   = (float*)carve(16 * 256 * 4);
  float* cSt   = (float*)carve(16 * 256 * 4);
  float* encmem= (float*)carve((size_t)8 * 45 * 256 * 4);
  float* Henc  = (float*)carve((size_t)8 * 45 * 256 * 4);
  float* yi    = (float*)carve(64 * 4);
  float* cCtx  = (float*)carve(2048 * 4);
  float* h0    = (float*)carve(2048 * 4);
  float* c0    = (float*)carve(2048 * 4);
  float* h1    = (float*)carve(2048 * 4);
  float* c1    = (float*)carve(2048 * 4);
  float* ysbuf = (float*)carve(1920 * 4);

  auto zero = [&](float* p, int n) {
    zero_k<<<dim3((n + 255) / 256), dim3(256), 0, stream>>>(p, n);
  };
  auto packW = [&](const Lw& w, __bf16* Bp, float* bias, int in_dim, int Kt) {
    int tot = Kt * 64 * 512;
    pack_B_k<<<dim3((tot + 255) / 256), dim3(256), 0, stream>>>(w.Wih, w.Whh, Bp, in_dim, 256, Kt);
    bias_k<<<dim3(4), dim3(256), 0, stream>>>(w.bih, w.bhh, bias);
  };
  auto gemm = [&](const __bf16* A, const __bf16* Bp, const float* bias, float* C, int Mt, int Kt) {
    int waves = ((Mt + 1) / 2) * (64 / 4);   // 2x4 super-tiles
    int thr = waves * 32;
    gemm_k<<<dim3((thr + 255) / 256), dim3(256), 0, stream>>>(A, Bp, bias, C, Mt, 64, Kt);
  };

  // ---- pack all weights to fragment-ordered bf16 (once per call) ----
  packW(enc, encB, encBias, FEAT, 10);                 // K = 43+256 -> 320
  for (int i = 0; i < 6; ++i) packW(pyr[i], pyrB[i], pyrBias[i], 512, 24);  // K = 768
  packW(dec0, dec0B, dec0Bias, 264, 17);               // K = 264+256 -> 544
  packW(dec1, dec1B, dec1Bias, 256, 16);               // K = 512

  // ---- encoder: 48 sequential WMMA steps over 2880 rows ----
  zero(hEnc, Nn * 256); zero(cEnc, Nn * 256);
  for (int t = 0; t < Tt; ++t) {
    int totA = 180 * 10 * 512;
    pack_A_k<<<dim3((totA + 255) / 256), dim3(256), 0, stream>>>(
        xs + (size_t)t * Nn * FEAT, FEAT, FEAT, nullptr, 0, 0,
        hEnc, 256, Abf, Nn, 180, 256, 10);
    gemm(Abf, encB, encBias, gates, 180, 10);
    lstm_cell_k<<<dim3((Nn * 256 + 255) / 256), dim3(256), 0, stream>>>(
        gates, hEnc, cEnc, hlast, xlengths, t, Nn);
  }
  reshape_hlast_k<<<dim3(8 * 180 * 512 / 256), dim3(256), 0, stream>>>(hlast, seq1);

  // ---- pyramid BiLSTM stages ----
  auto run_dir = [&](const float* seqp, int L, const __bf16* Bp, const float* bias,
                     float* hs, bool rev) {
    zero(hSt, 16 * 256); zero(cSt, 16 * 256);
    for (int s = 0; s < L; ++s) {
      int t = rev ? (L - 1 - s) : s;
      int totA = 1 * 24 * 512;
      pack_A_k<<<dim3((totA + 255) / 256), dim3(256), 0, stream>>>(
          seqp + (size_t)t * 512, (size_t)L * 512, 512, nullptr, 0, 0,
          hSt, 256, Abf, 8, 1, 256, 24);
      gemm(Abf, Bp, bias, gates, 1, 24);
      lstm_cell_k<<<dim3(8), dim3(256), 0, stream>>>(
          gates, hSt, cSt, hs + (size_t)t * 8 * 256, nullptr, 0, 8);
    }
  };
  run_dir(seq1, 180, pyrB[0], pyrBias[0], hsF, false);
  run_dir(seq1, 180, pyrB[1], pyrBias[1], hsB, true);
  pyr_combine_k<<<dim3(8 * 90 * 512 / 256), dim3(256), 0, stream>>>(hsF, hsB, seq2, 180);
  run_dir(seq2, 90, pyrB[2], pyrBias[2], hsF, false);
  run_dir(seq2, 90, pyrB[3], pyrBias[3], hsB, true);
  pyr_combine_k<<<dim3(8 * 45 * 512 / 256), dim3(256), 0, stream>>>(hsF, hsB, seq3, 90);
  run_dir(seq3, 45, pyrB[4], pyrBias[4], hsF, false);
  run_dir(seq3, 45, pyrB[5], pyrBias[5], hsB, true);
  avg_combine_k<<<dim3(8 * 45 * 256 / 256), dim3(256), 0, stream>>>(hsF, hsB, encmem, 45);

  // ---- LayerNorm + tanh MLP -> attention memory ----
  ln_mlp_k<<<dim3(360), dim3(256), 0, stream>>>(encmem, gamma, beta, W1, b1, Henc);

  // ---- decoder: 30 steps of 2-layer LSTM + dot attention ----
  zero(yi, 64); zero(cCtx, 2048);
  zero(h0, 2048); zero(c0, 2048); zero(h1, 2048); zero(c1, 2048);
  for (int s = 0; s < 30; ++s) {
    int totA0 = 17 * 512;
    pack_A_k<<<dim3((totA0 + 255) / 256), dim3(256), 0, stream>>>(
        yi, 8, 8, cCtx, 256, 256, h0, 256, Abf, 8, 1, 256, 17);
    gemm(Abf, dec0B, dec0Bias, gates, 1, 17);
    lstm_cell_k<<<dim3(8), dim3(256), 0, stream>>>(gates, h0, c0, nullptr, nullptr, 0, 8);
    int totA1 = 16 * 512;
    pack_A_k<<<dim3((totA1 + 255) / 256), dim3(256), 0, stream>>>(
        h0, 256, 256, nullptr, 0, 0, h1, 256, Abf, 8, 1, 256, 16);
    gemm(Abf, dec1B, dec1Bias, gates, 1, 16);
    lstm_cell_k<<<dim3(8), dim3(256), 0, stream>>>(gates, h1, c1, nullptr, nullptr, 0, 8);
    dec_attn_k<<<dim3(1), dim3(256), 0, stream>>>(h1, W2, b2, Henc, Wo, bo,
                                                  yi, cCtx, ysbuf + (size_t)s * 64);
  }
  final_k<<<dim3((1920 + 255) / 256), dim3(256), 0, stream>>>(ysbuf, out);
}
